// DQN_31310311587959
// MI455X (gfx1250) — matrix operand order repaired and checked by reading, not compiled, not run on hardware
//
#include <hip/hip_runtime.h>
#include <hip/hip_bf16.h>
#include <cstdint>

// ---------------------------------------------------------------------------
// CDNA5 (gfx1250) wave32 WMMA GNN pipeline, LDS-staged.
// ---------------------------------------------------------------------------

typedef _Float16 v8h  __attribute__((ext_vector_type(8)));
typedef _Float16 v16h __attribute__((ext_vector_type(16)));
typedef float    v8f  __attribute__((ext_vector_type(8)));

static __device__ __forceinline__ v8f wmma_f16(v16h a, v16h b, v8f c) {
  // D = A(16x32 f16) * B(32x16 f16) + C(16x16 f32)
  return __builtin_amdgcn_wmma_f32_16x16x32_f16(
      /*neg_a=*/false, a, /*neg_b=*/false, b,
      /*c_mod=*/(short)0, c, /*reuse_a=*/false, /*reuse_b=*/false);
}

static __device__ __forceinline__ v8f zero8() {
  v8f z;
#pragma unroll
  for (int i = 0; i < 8; ++i) z[i] = 0.0f;
  return z;
}

static __device__ __forceinline__ v16h cat8(v8h lo, v8h hi) {
  return __builtin_shufflevector(lo, hi, 0, 1, 2, 3, 4, 5, 6, 7,
                                         8, 9, 10, 11, 12, 13, 14, 15);
}

// float atomic-max via signed-int max / unsigned-int min bit trick (no NaNs).
static __device__ __forceinline__ void atomicMaxFloat(float* addr, float val) {
  if (val >= 0.0f) {
    atomicMax((int*)addr, __float_as_int(val));
  } else {
    atomicMin((unsigned int*)addr, __float_as_uint(val));
  }
}

// ---------------------------------------------------------------------------
// Pack a row-major f32 weight [K x 64] into WMMA B-fragment order, f16:
// layout [kb][nb][lane][16 halves]; lane<16: N=nb*16+lane, K=kb*32+(e&7)+16*(e>>3);
// lane>=16: N=nb*16+lane-16, K=kb*32+8+(e&7)+16*(e>>3).  (ISA 7.12.2 16-bit B layout)
// ---------------------------------------------------------------------------
__global__ void pack_bfrag_kernel(const float* __restrict__ src,
                                  _Float16* __restrict__ dst, int K) {
  int p = blockIdx.x * blockDim.x + threadIdx.x;
  int total = K * 64;
  if (p >= total) return;
  int kb   = p >> 11;        // / 2048  (one 32x64 K-chunk)
  int rem  = p & 2047;
  int nb   = rem >> 9;       // / 512
  int rem2 = rem & 511;
  int lane = rem2 >> 4;
  int e    = rem2 & 15;
  int krow = kb * 32 + ((lane >> 4) << 3) + (e & 7) + ((e >> 3) << 4);
  int ncol = nb * 16 + (lane & 15);
  dst[p] = (_Float16)src[krow * 64 + ncol];
}

// agg (n64 floats) = 0 ; aggm (2*n64 floats) = -inf
__global__ void init_accum_kernel(float* __restrict__ agg,
                                  float* __restrict__ aggm, long n64) {
  long i = (long)blockIdx.x * blockDim.x + threadIdx.x;
  if (i < n64) agg[i] = 0.0f;
  long j = i - n64;
  if (j >= 0 && j < 2 * n64) aggm[j] = -__builtin_inff();
}

__global__ void zero_f16_kernel(_Float16* __restrict__ p, long n) {
  long i = (long)blockIdx.x * blockDim.x + threadIdx.x;
  if (i < n) p[i] = (_Float16)0.0f;
}

// n = relu(x @ n_enc_w + n_enc_b)  -> f16 [N,64]
__global__ void node_enc_kernel(const float* __restrict__ x,
                                const float* __restrict__ w,
                                const float* __restrict__ b,
                                _Float16* __restrict__ n16, long n64) {
  long i = (long)blockIdx.x * blockDim.x + threadIdx.x;
  if (i >= n64) return;
  long node = i >> 6;
  int col = (int)(i & 63);
  float acc = b[col] + x[node * 3 + 0] * w[0 * 64 + col]
                     + x[node * 3 + 1] * w[1 * 64 + col]
                     + x[node * 3 + 2] * w[2 * 64 + col];
  n16[i] = (_Float16)fmaxf(acc, 0.0f);
}

// ---------------------------------------------------------------------------
// conv1 fused edge kernel: one wave per 16-edge tile, 8 waves/block.
//  - block stages packed c1_rel B-fragments (24KB) in LDS
//  - each wave gathers its 16x96 A-tile (n16[src] || edge-enc) into LDS
//  - 24 WMMAs per tile (2 relations x 3 K-steps x 4 N-blocks)
//  - per-row relation select, atomicAdd scatter into agg[dst]
// ---------------------------------------------------------------------------
__global__ __launch_bounds__(256)
void conv1_edge_kernel(const _Float16* __restrict__ n16,
                       const float* __restrict__ ea,      // [E,2]
                       const float* __restrict__ ew,      // [2,32]
                       const float* __restrict__ eb,      // [32]
                       const _Float16* __restrict__ pkrel,// [2][3][4][32][16]
                       const int* __restrict__ src,
                       const int* __restrict__ etype,
                       const int* __restrict__ dst,
                       float* __restrict__ agg,
                       int numTiles, int E) {
  __shared__ __attribute__((aligned(16))) _Float16 sB[12288];    // 24KB
  __shared__ __attribute__((aligned(16))) _Float16 sF[8][1536];  // 24KB (16x96 / wave)

  const int tid   = threadIdx.x;
  const int w     = tid >> 5;          // wave in block
  const int lane  = tid & 31;
  const int tile  = blockIdx.x * 8 + w;
  const bool active = tile < numTiles;

  // stage B fragments: 12288 halves = 1536 x v8h, 256 threads x 6 iters
#pragma unroll
  for (int it = 0; it < 6; ++it) {
    int t = tid + it * 256;
    ((v8h*)sB)[t] = ((const v8h*)pkrel)[t];
  }
  __syncthreads();

  // gather A tile: 2 lanes per edge row; lane covers 32 node-halves + 16 enc
  if (active) {
    int erow = lane >> 1;              // 0..15
    int part = lane & 1;               // 0/1
    int e = tile * 16 + erow;
    _Float16* frow = &sF[w][erow * 96];
    if (e < E) {
      const _Float16* nrow = n16 + (size_t)src[e] * 64 + part * 32;
#pragma unroll
      for (int q = 0; q < 4; ++q)
        *(v8h*)(frow + part * 32 + q * 8) = *(const v8h*)(nrow + q * 8);
      float a0 = ea[(size_t)e * 2 + 0];
      float a1 = ea[(size_t)e * 2 + 1];
#pragma unroll
      for (int j = 0; j < 16; ++j) {
        int col = part * 16 + j;
        float v = eb[col] + a0 * ew[col] + a1 * ew[32 + col];
        frow[64 + col] = (_Float16)fmaxf(v, 0.0f);
      }
    } else {
#pragma unroll
      for (int q = 0; q < 6; ++q)
        *(v8h*)(frow + part * 32 + q * 8) = (v8h)0;   // zero node part
#pragma unroll
      for (int j = 0; j < 16; ++j)
        frow[64 + part * 16 + j] = (_Float16)0.0f;
    }
  }
  __syncthreads();
  if (!active) return;

  v8f acc[2][4];
#pragma unroll
  for (int r = 0; r < 2; ++r)
#pragma unroll
    for (int nb = 0; nb < 4; ++nb) acc[r][nb] = zero8();

  const int row = lane & 15;
  const int hi  = lane >> 4;
  const _Float16* rowp = &sF[w][row * 96 + hi * 8];

#pragma unroll
  for (int kb = 0; kb < 3; ++kb) {
    v8h lo = *(const v8h*)(rowp + kb * 32);
    v8h hh = *(const v8h*)(rowp + kb * 32 + 16);
    v16h a = cat8(lo, hh);
#pragma unroll
    for (int r = 0; r < 2; ++r) {
#pragma unroll
      for (int nb = 0; nb < 4; ++nb) {
        v16h bf = *(const v16h*)(sB + (size_t)(((r * 3 + kb) * 4 + nb) * 32 + lane) * 16);
        acc[r][nb] = wmma_f16(a, bf, acc[r][nb]);
      }
    }
  }

  const int ncol = lane & 15;
#pragma unroll
  for (int i = 0; i < 8; ++i) {
    int e = tile * 16 + i + hi * 8;
    if (e < E) {
      int rt = etype[e];
      float* base = agg + (size_t)dst[e] * 64;
#pragma unroll
      for (int nb = 0; nb < 4; ++nb) {
        float v = rt ? acc[1][nb][i] : acc[0][nb][i];
        atomicAdd(base + nb * 16 + ncol, v);
      }
    }
  }
}

// h = relu(n @ c1_root + agg + c1_bias) -> f16 ; one wave per 16-node tile.
__global__ __launch_bounds__(256)
void conv1_node_kernel(const _Float16* __restrict__ n16,
                       const _Float16* __restrict__ pkroot,  // [2][4][32][16]
                       const float* __restrict__ agg,
                       const float* __restrict__ bias,
                       _Float16* __restrict__ h16,
                       int numTiles, int N) {
  int wave = (blockIdx.x * blockDim.x + threadIdx.x) >> 5;
  int lane = threadIdx.x & 31;
  if (wave >= numTiles) return;

  v8f acc[4];
#pragma unroll
  for (int nb = 0; nb < 4; ++nb) acc[nb] = zero8();

  int row = lane & 15;
  int hi  = lane >> 4;
  const _Float16* rowp = n16 + ((size_t)wave * 16 + row) * 64 + hi * 8;

#pragma unroll
  for (int kb = 0; kb < 2; ++kb) {
    v16h a = cat8(*(const v8h*)(rowp + kb * 32),
                  *(const v8h*)(rowp + kb * 32 + 16));
#pragma unroll
    for (int nb = 0; nb < 4; ++nb) {
      v16h bf = *(const v16h*)(pkroot + (size_t)((kb * 4 + nb) * 32 + lane) * 16);
      acc[nb] = wmma_f16(a, bf, acc[nb]);
    }
  }

  int ncol = lane & 15;
#pragma unroll
  for (int i = 0; i < 8; ++i) {
    int node = wave * 16 + i + hi * 8;
    if (node < N) {
#pragma unroll
      for (int nb = 0; nb < 4; ++nb) {
        int col = nb * 16 + ncol;
        float v = acc[nb][i] + agg[(size_t)node * 64 + col] + bias[col];
        h16[(size_t)node * 64 + col] = (_Float16)fmaxf(v, 0.0f);
      }
    }
  }
}

// out_pre = h@c2_root + c2_bias (f32) ; tr[r] = h@c2_rel[r] (f16, halves gather
// traffic for the edge-max pass).  One wave per 16-node tile.
__global__ __launch_bounds__(256)
void conv2_node_kernel(const _Float16* __restrict__ h16,
                       const _Float16* __restrict__ pkroot,  // [2][4][32][16]
                       const _Float16* __restrict__ pkrel,   // [2][2][4][32][16]
                       const float* __restrict__ bias,
                       float* __restrict__ out_pre,
                       _Float16* __restrict__ tr,            // [2][N][64] f16
                       int numTiles, int N, long N64) {
  int wave = (blockIdx.x * blockDim.x + threadIdx.x) >> 5;
  int lane = threadIdx.x & 31;
  if (wave >= numTiles) return;

  const _Float16* bases[3] = { pkroot, pkrel, pkrel + 4096 };

  v8f acc[3][4];
#pragma unroll
  for (int w = 0; w < 3; ++w)
#pragma unroll
    for (int nb = 0; nb < 4; ++nb) acc[w][nb] = zero8();

  int row = lane & 15;
  int hi  = lane >> 4;
  const _Float16* rowp = h16 + ((size_t)wave * 16 + row) * 64 + hi * 8;

#pragma unroll
  for (int kb = 0; kb < 2; ++kb) {
    v16h a = cat8(*(const v8h*)(rowp + kb * 32),
                  *(const v8h*)(rowp + kb * 32 + 16));
#pragma unroll
    for (int w = 0; w < 3; ++w) {
#pragma unroll
      for (int nb = 0; nb < 4; ++nb) {
        v16h bf = *(const v16h*)(bases[w] + (size_t)((kb * 4 + nb) * 32 + lane) * 16);
        acc[w][nb] = wmma_f16(a, bf, acc[w][nb]);
      }
    }
  }

  int ncol = lane & 15;
#pragma unroll
  for (int i = 0; i < 8; ++i) {
    int node = wave * 16 + i + hi * 8;
    if (node < N) {
#pragma unroll
      for (int nb = 0; nb < 4; ++nb) {
        int col = nb * 16 + ncol;
        size_t idx = (size_t)node * 64 + col;
        out_pre[idx]  = acc[0][nb][i] + bias[col];
        tr[idx]       = (_Float16)acc[1][nb][i];
        tr[N64 + idx] = (_Float16)acc[2][nb][i];
      }
    }
  }
}

// per-relation segment max:  aggm[r][dst] = max(aggm, tr[r][src])
// grid: <<<E, 64>>>
__global__ void conv2_edge_kernel(const _Float16* __restrict__ tr,
                                  const int* __restrict__ src,
                                  const int* __restrict__ dstp,
                                  const int* __restrict__ etype,
                                  float* __restrict__ aggm, long N64) {
  long e = blockIdx.x;
  int col = threadIdx.x;
  int r = etype[e];
  float v = (float)tr[(size_t)r * N64 + (size_t)src[e] * 64 + col];
  atomicMaxFloat(aggm + (size_t)r * N64 + (size_t)dstp[e] * 64 + col, v);
}

// h2 = relu(out_pre + finite(aggm0) + finite(aggm1)); dueling heads.
// One wave per node; wave32 shuffle reduction.
__global__ __launch_bounds__(256)
void final_kernel(const float* __restrict__ out_pre,
                  const float* __restrict__ aggm,
                  const float* __restrict__ v_w, const float* __restrict__ v_b,
                  const float* __restrict__ a_w, const float* __restrict__ a_b,
                  float* __restrict__ out, long N64, int N) {
  int node = (blockIdx.x * blockDim.x + threadIdx.x) >> 5;
  int lane = threadIdx.x & 31;
  if (node >= N) return;

  float pv = 0.0f;
  float pa[5] = {0.0f, 0.0f, 0.0f, 0.0f, 0.0f};
#pragma unroll
  for (int t = 0; t < 2; ++t) {
    int col = lane + t * 32;
    size_t idx = (size_t)node * 64 + col;
    float m0 = aggm[idx];
    float m1 = aggm[N64 + idx];
    float h2 = out_pre[idx] +
               (m0 > -__builtin_inff() ? m0 : 0.0f) +
               (m1 > -__builtin_inff() ? m1 : 0.0f);
    h2 = fmaxf(h2, 0.0f);
    pv += h2 * v_w[col];
#pragma unroll
    for (int o = 0; o < 5; ++o) pa[o] += h2 * a_w[col * 5 + o];
  }
#pragma unroll
  for (int off = 16; off >= 1; off >>= 1) {
    pv += __shfl_xor(pv, off, 32);
#pragma unroll
    for (int o = 0; o < 5; ++o) pa[o] += __shfl_xor(pa[o], off, 32);
  }
  if (lane == 0) {
    out[node] = pv + v_b[0];
#pragma unroll
    for (int o = 0; o < 5; ++o)
      out[(size_t)N + (size_t)node * 5 + o] = pa[o] + a_b[o];
  }
}

// ---------------------------------------------------------------------------
extern "C" void kernel_launch(void* const* d_in, const int* in_sizes, int n_in,
                              void* d_out, int out_size, void* d_ws, size_t ws_size,
                              hipStream_t stream) {
  const float* x         = (const float*)d_in[0];
  const float* edge_attr = (const float*)d_in[1];
  const int*   eidx      = (const int*)d_in[2];
  const int*   etype     = (const int*)d_in[3];
  const float* n_enc_w   = (const float*)d_in[4];
  const float* n_enc_b   = (const float*)d_in[5];
  const float* e_enc_w   = (const float*)d_in[6];
  const float* e_enc_b   = (const float*)d_in[7];
  const float* c1_root   = (const float*)d_in[8];
  const float* c1_rel    = (const float*)d_in[9];
  const float* c1_bias   = (const float*)d_in[10];
  const float* c2_root   = (const float*)d_in[11];
  const float* c2_rel    = (const float*)d_in[12];
  const float* c2_bias   = (const float*)d_in[13];
  const float* v_w       = (const float*)d_in[14];
  const float* v_b       = (const float*)d_in[15];
  const float* a_w       = (const float*)d_in[16];
  const float* a_b       = (const float*)d_in[17];

  const int N = in_sizes[0] / 3;
  const int E = in_sizes[3];
  const int* src  = eidx;
  const int* dstp = eidx + E;
  const long N64 = (long)N * 64;

  const int tilesE = (E + 15) / 16;
  const int tilesN = (N + 15) / 16;
  const long Npad = (long)tilesN * 16;

  // ---- workspace carve-out (256B aligned) ----
  char* wsb = (char*)d_ws;
  size_t off = 0;
  auto carve = [&](size_t bytes) -> char* {
    char* p = wsb + off;
    off = (off + bytes + 255) & ~(size_t)255;
    return p;
  };
  _Float16* n16      = (_Float16*)carve((size_t)Npad * 64 * 2);
  float*    agg      = (float*)   carve((size_t)N64 * 4);
  _Float16* h16      = (_Float16*)carve((size_t)Npad * 64 * 2);
  float*    out_pre  = (float*)   carve((size_t)N64 * 4);
  _Float16* tr       = (_Float16*)carve((size_t)2 * N64 * 2);
  float*    aggm     = (float*)   carve((size_t)2 * N64 * 4);
  _Float16* pk_c1rel = (_Float16*)carve((size_t)2 * 96 * 64 * 2);
  _Float16* pk_c1rt  = (_Float16*)carve((size_t)64 * 64 * 2);
  _Float16* pk_c2rt  = (_Float16*)carve((size_t)64 * 64 * 2);
  _Float16* pk_c2rel = (_Float16*)carve((size_t)2 * 64 * 64 * 2);
  (void)ws_size; (void)n_in; (void)out_size;

  const int TPB = 256;

  // 1) weight packing into WMMA B-fragment order
  pack_bfrag_kernel<<<(96 * 64 + TPB - 1) / TPB, TPB, 0, stream>>>(c1_rel,            pk_c1rel,        96);
  pack_bfrag_kernel<<<(96 * 64 + TPB - 1) / TPB, TPB, 0, stream>>>(c1_rel + 96 * 64,  pk_c1rel + 6144, 96);
  pack_bfrag_kernel<<<(64 * 64 + TPB - 1) / TPB, TPB, 0, stream>>>(c1_root,           pk_c1rt,         64);
  pack_bfrag_kernel<<<(64 * 64 + TPB - 1) / TPB, TPB, 0, stream>>>(c2_root,           pk_c2rt,         64);
  pack_bfrag_kernel<<<(64 * 64 + TPB - 1) / TPB, TPB, 0, stream>>>(c2_rel,            pk_c2rel,        64);
  pack_bfrag_kernel<<<(64 * 64 + TPB - 1) / TPB, TPB, 0, stream>>>(c2_rel + 64 * 64,  pk_c2rel + 4096, 64);

  // 2) init accumulators + padded activation tail
  {
    long tot = 3 * N64;
    init_accum_kernel<<<(int)((tot + TPB - 1) / TPB), TPB, 0, stream>>>(agg, aggm, N64);
  }
  if (Npad > N) {
    long padElems = (Npad - N) * 64;
    zero_f16_kernel<<<(int)((padElems + TPB - 1) / TPB), TPB, 0, stream>>>(n16 + (size_t)N * 64, padElems);
    zero_f16_kernel<<<(int)((padElems + TPB - 1) / TPB), TPB, 0, stream>>>(h16 + (size_t)N * 64, padElems);
  }

  // 3) node encoder
  node_enc_kernel<<<(int)((N64 + TPB - 1) / TPB), TPB, 0, stream>>>(x, n_enc_w, n_enc_b, n16, N64);

  // 4) conv1: fused gather + WMMA messages + scatter-sum, then root combine
  conv1_edge_kernel<<<(tilesE + 7) / 8, TPB, 0, stream>>>(
      n16, edge_attr, e_enc_w, e_enc_b, pk_c1rel, src, etype, dstp, agg, tilesE, E);
  conv1_node_kernel<<<(tilesN * 32 + TPB - 1) / TPB, TPB, 0, stream>>>(
      n16, pk_c1rt, agg, c1_bias, h16, tilesN, N);

  // 5) conv2: node transforms (root + 2 relations), then scatter-max
  conv2_node_kernel<<<(tilesN * 32 + TPB - 1) / TPB, TPB, 0, stream>>>(
      h16, pk_c2rt, pk_c2rel, c2_bias, out_pre, tr, tilesN, N, N64);
  conv2_edge_kernel<<<E, 64, 0, stream>>>(tr, src, dstp, etype, aggm, N64);

  // 6) finalize: combine, relu, dueling heads
  final_kernel<<<(N * 32 + TPB - 1) / TPB, TPB, 0, stream>>>(
      out_pre, aggm, v_w, v_b, a_w, a_b, (float*)d_out, N64, N);
}
